// TextModels_32787780337869
// MI455X (gfx1250) — matrix operand order repaired and checked by reading, not compiled, not run on hardware
//
#include <hip/hip_runtime.h>
#include <hip/hip_bf16.h>

// ---------------------------------------------------------------------------
// Hierarchical attention net on gfx1250: all GEMMs via v_wmma_f32_16x16x32_bf16,
// GRU input staging via double-buffered global_load_async_to_lds_b128.
// S=32 sentences, T=64 tokens, B=64 batch, E=H=G=128, C=5
// ---------------------------------------------------------------------------

typedef __attribute__((ext_vector_type(16))) unsigned short v16u;
typedef __attribute__((ext_vector_type(16))) __bf16         v16bf;
typedef __attribute__((ext_vector_type(8)))  float          v8f;

union FragU { v16u v; uint4 q[2]; };
union Pack4 { unsigned short u[4]; uint2 v; };

__device__ __forceinline__ unsigned short f2b(float x) {
  unsigned int u = __builtin_bit_cast(unsigned int, x);
  u = (u + 0x7FFFu + ((u >> 16) & 1u)) >> 16;   // round-to-nearest-even
  return (unsigned short)u;
}
__device__ __forceinline__ float b2f(unsigned short h) {
  unsigned int u = ((unsigned int)h) << 16;
  return __builtin_bit_cast(float, u);
}

__device__ __forceinline__ float fast_tanh(float x) {
#if defined(__has_builtin) && __has_builtin(__builtin_amdgcn_tanhf)
  return __builtin_amdgcn_tanhf(x);            // v_tanh_f32
#else
  float e = __expf(-2.f * fabsf(x));
  float t = (1.f - e) * __builtin_amdgcn_rcpf(1.f + e);
  return copysignf(t, x);
#endif
}
__device__ __forceinline__ float fast_sigmoid(float x) {
  return __builtin_amdgcn_rcpf(1.f + __expf(-x));   // v_exp + v_rcp
}

__device__ __forceinline__ v8f wmma_bf16(v16u a, v16u b, v8f c) {
  return __builtin_amdgcn_wmma_f32_16x16x32_bf16(
      false, __builtin_bit_cast(v16bf, a),
      false, __builtin_bit_cast(v16bf, b),
      (short)0, c, false, false);
}

// A-fragment 16x32 bf16 from row-major [rows][ldk] (ISA 7.12.2 16-bit A layout)
__device__ __forceinline__ v16u frag_a(const unsigned short* base, int ldk,
                                       int r0, int k0) {
  const int lane = threadIdx.x & 31;
  const unsigned short* p = base + (r0 + (lane & 15)) * ldk + k0 + ((lane >> 4) << 3);
  FragU f;
  f.q[0] = *(const uint4*)(p);
  f.q[1] = *(const uint4*)(p + 16);
  return f.v;
}

// B-fragment 32x16 bf16 from weights stored row-major W[n][k] (n = output col)
__device__ __forceinline__ v16u frag_b(const unsigned short* base, int ldk,
                                       int n0, int k0) {
  const int lane = threadIdx.x & 31;
  const unsigned short* p = base + (n0 + (lane & 15)) * ldk + k0 + ((lane >> 4) << 3);
  FragU f;
  f.q[0] = *(const uint4*)(p);
  f.q[1] = *(const uint4*)(p + 16);
  return f.v;
}

// async 16B global -> LDS copy (ASYNCcnt-tracked, ISA §10 / 08_async_tensor)
__device__ __forceinline__ void async_ld_b128(unsigned int lds_addr, const void* g) {
  asm volatile("global_load_async_to_lds_b128 %0, %1, off"
               :: "v"(lds_addr), "v"(g) : "memory");
}
__device__ __forceinline__ void wait_async0() {
  asm volatile("s_wait_asynccnt 0" ::: "memory");
}

// ---------------------------------------------------------------------------
// small prep kernels
// ---------------------------------------------------------------------------
__global__ void cvt_kernel(const float* __restrict__ src,
                           unsigned short* __restrict__ dst, int n) {
  int i = blockIdx.x * 256 + threadIdx.x;
  if (i < n) dst[i] = f2b(src[i]);
}

__global__ void transp256_kernel(const float* __restrict__ src,
                                 unsigned short* __restrict__ dst) {
  int i = blockIdx.x * 256 + threadIdx.x;   // i = n*256 + k
  int n = i >> 8, k = i & 255;
  dst[i] = f2b(src[k * 256 + n]);
}

// embedding gather, 8 elements (16B out) per thread
__global__ void embed_kernel(const int* __restrict__ tokens,
                             const float* __restrict__ table,
                             unsigned short* __restrict__ emb) {
  int i = blockIdx.x * 256 + threadIdx.x;   // over S*T*B*E/8 = 2,097,152
  int e8 = (i & 15) << 3, rb = i >> 4;
  int tok = tokens[rb];
  const float4* src = (const float4*)(table + tok * 128 + e8);
  float4 a = src[0], b = src[1];
  Pack4 lo, hi;
  lo.u[0] = f2b(a.x); lo.u[1] = f2b(a.y); lo.u[2] = f2b(a.z); lo.u[3] = f2b(a.w);
  hi.u[0] = f2b(b.x); hi.u[1] = f2b(b.y); hi.u[2] = f2b(b.z); hi.u[3] = f2b(b.w);
  uint4 o; o.x = lo.v.x; o.y = lo.v.y; o.z = hi.v.x; o.w = hi.v.y;
  *(uint4*)(emb + (size_t)rb * 128 + e8) = o;
}

// ---------------------------------------------------------------------------
// biGRU: one workgroup per (sequence, direction); recurrence lives in LDS.
// 8 waves x 3 n-tiles cover 3H=384 gate outputs; 4 m-tiles cover 64 rows.
// x_t staging is double-buffered via async global->LDS loads.
// ---------------------------------------------------------------------------
template <int T_STEPS, int KX>
__global__ __launch_bounds__(256)
void gru_kernel(const unsigned short* __restrict__ x_all,  // [(nseq*T)][64][KX]
                const unsigned short* __restrict__ Whh,    // [2][384][128] bf16
                const unsigned short* __restrict__ Wih,    // [2][384][KX] bf16
                const float* __restrict__ bih,             // [2][384]
                const float* __restrict__ bhh,             // [2][384]
                unsigned short* __restrict__ out)          // [(nseq*T)][64][256]
{
  __shared__ float          rz_s[64][256];     // 64 KB  r|z pre-activations
  __shared__ float          hn_s[64][128];     // 32 KB  gh n-part
  __shared__ float          inn_s[64][128];    // 32 KB  xg n-part
  __shared__ float          h_f[64][128];      // 32 KB  fp32 hidden state
  __shared__ unsigned short h_b[64][128];      // 16 KB  bf16 hidden (WMMA A)
  __shared__ unsigned short x_b[2][64][KX];    // <=64 KB double-buffered input

  const int s    = blockIdx.x;
  const int dir  = blockIdx.y;
  const int tid  = threadIdx.x;
  const int w    = tid >> 5;
  const int lane = tid & 31;
  constexpr int KXI = KX / 32;      // WMMA k-iterations for input GEMM
  constexpr int CH  = KX / 32;      // 16B async chunks per thread per step

  for (int i = tid; i < 64 * 128; i += 256) {
    h_f[i >> 7][i & 127] = 0.f;
    h_b[i >> 7][i & 127] = 0;
  }

  const unsigned short* Wh = Whh + dir * 384 * 128;
  const unsigned short* Wx = Wih + (size_t)dir * 384 * KX;
  const float* bi = bih + dir * 384;
  const float* bh = bhh + dir * 384;

  // issue async stage of step 0 into buffer 0
  {
    const int t0 = dir ? (T_STEPS - 1) : 0;
    const char* g = (const char*)(x_all + ((size_t)(s * T_STEPS + t0)) * 64 * KX);
    unsigned lb = (unsigned)(uintptr_t)(&x_b[0][0][0]);
#pragma unroll
    for (int c = 0; c < CH; ++c) {
      unsigned off = (unsigned)(tid + c * 256) * 16u;
      async_ld_b128(lb + off, g + off);
    }
  }

  for (int step = 0; step < T_STEPS; ++step) {
    const int t_in = dir ? (T_STEPS - 1 - step) : step;
    const int buf  = step & 1;

    wait_async0();       // our async chunks of x_b[buf] landed
    __syncthreads();     // everyone's chunks + previous-step h_b visible

    // prefetch next step's input into the other buffer (overlaps the GEMM)
    if (step + 1 < T_STEPS) {
      const int t_nx = dir ? (T_STEPS - 2 - step) : (step + 1);
      const char* g = (const char*)(x_all + ((size_t)(s * T_STEPS + t_nx)) * 64 * KX);
      unsigned lb = (unsigned)(uintptr_t)(&x_b[buf ^ 1][0][0]);
#pragma unroll
      for (int c = 0; c < CH; ++c) {
        unsigned off = (unsigned)(tid + c * 256) * 16u;
        async_ld_b128(lb + off, g + off);
      }
    }

    // ---- gh = h_b x Whh^T (K=128) ----
#pragma unroll
    for (int j = 0; j < 3; ++j) {
      const int n0 = (w * 3 + j) * 16;
      v16u Bf[4];
#pragma unroll
      for (int k = 0; k < 4; ++k) Bf[k] = frag_b(Wh, 128, n0, k * 32);
#pragma unroll
      for (int mt = 0; mt < 4; ++mt) {
        v8f acc = {0.f, 0.f, 0.f, 0.f, 0.f, 0.f, 0.f, 0.f};
#pragma unroll
        for (int k = 0; k < 4; ++k)
          acc = wmma_bf16(frag_a(&h_b[0][0], 128, mt * 16, k * 32), Bf[k], acc);
        const int col = n0 + (lane & 15);
        const int r0  = mt * 16 + ((lane >> 4) << 3);
#pragma unroll
        for (int jj = 0; jj < 8; ++jj) {
          if (n0 < 256) rz_s[r0 + jj][col]       = acc[jj];
          else          hn_s[r0 + jj][col - 256] = acc[jj];
        }
      }
    }
    // ---- xg = x_b x Wih^T (K=KX); same lane->element mapping, so += is safe ----
#pragma unroll
    for (int j = 0; j < 3; ++j) {
      const int n0 = (w * 3 + j) * 16;
      v16u Bf[KXI];
#pragma unroll
      for (int k = 0; k < KXI; ++k) Bf[k] = frag_b(Wx, KX, n0, k * 32);
#pragma unroll
      for (int mt = 0; mt < 4; ++mt) {
        v8f acc = {0.f, 0.f, 0.f, 0.f, 0.f, 0.f, 0.f, 0.f};
#pragma unroll
        for (int k = 0; k < KXI; ++k)
          acc = wmma_bf16(frag_a(&x_b[buf][0][0], KX, mt * 16, k * 32), Bf[k], acc);
        const int col = n0 + (lane & 15);
        const int r0  = mt * 16 + ((lane >> 4) << 3);
#pragma unroll
        for (int jj = 0; jj < 8; ++jj) {
          if (n0 < 256) rz_s[r0 + jj][col]        += acc[jj];
          else          inn_s[r0 + jj][col - 256]  = acc[jj];
        }
      }
    }
    __syncthreads();

    // ---- gates (fp32, 4 consecutive units per thread), update h, emit bf16 ----
    for (int i0 = tid * 4; i0 < 64 * 128; i0 += 1024) {
      const int row = i0 >> 7, jc = i0 & 127;
      Pack4 hb;
      float hnew[4];
#pragma unroll
      for (int q = 0; q < 4; ++q) {
        const int j = jc + q;
        float r = fast_sigmoid(rz_s[row][j]       + bi[j]       + bh[j]);
        float z = fast_sigmoid(rz_s[row][128 + j] + bi[128 + j] + bh[128 + j]);
        float n = fast_tanh(inn_s[row][j] + bi[256 + j] + r * (hn_s[row][j] + bh[256 + j]));
        hnew[q] = (1.f - z) * n + z * h_f[row][j];
        h_f[row][j] = hnew[q];
        hb.u[q] = f2b(hnew[q]);
      }
      *(uint2*)&h_b[row][jc] = hb.v;
      *(uint2*)&out[(((size_t)(s * T_STEPS + t_in)) * 64 + row) * 256 + dir * 128 + jc] = hb.v;
    }
    __syncthreads();
  }
}

// ---------------------------------------------------------------------------
// fused attention logits: a[m] = sum_n tanh( (X W)[m,n] + bias[n] ) * proj[n]
// X: [M][256] bf16, WT: [256 n][256 k] bf16. One wave per 16-row tile.
// ---------------------------------------------------------------------------
__global__ __launch_bounds__(256)
void attn_kernel(const unsigned short* __restrict__ X,
                 const unsigned short* __restrict__ WT,
                 const float* __restrict__ bias,
                 const float* __restrict__ proj,
                 float* __restrict__ outv, int M)
{
  const int tid = threadIdx.x, w = tid >> 5, lane = tid & 31;
  const int m0 = (blockIdx.x * 8 + w) * 16;
  if (m0 >= M) return;

  v16u Af[8];
#pragma unroll
  for (int k = 0; k < 8; ++k) Af[k] = frag_a(X, 256, m0, k * 32);

  float sacc[8] = {0.f, 0.f, 0.f, 0.f, 0.f, 0.f, 0.f, 0.f};
#pragma unroll 1
  for (int nt = 0; nt < 16; ++nt) {
    v8f acc = {0.f, 0.f, 0.f, 0.f, 0.f, 0.f, 0.f, 0.f};
#pragma unroll
    for (int k = 0; k < 8; ++k)
      acc = wmma_bf16(Af[k], frag_b(WT, 256, nt * 16, k * 32), acc);
    const int col = nt * 16 + (lane & 15);
    const float b = bias[col], p = proj[col];
#pragma unroll
    for (int jj = 0; jj < 8; ++jj)
      sacc[jj] += fast_tanh(acc[jj] + b) * p;
  }
  // reduce across the 16 lanes of each half-wave (butterfly stays in-group)
#pragma unroll
  for (int jj = 0; jj < 8; ++jj) {
    float v = sacc[jj];
    for (int m = 8; m >= 1; m >>= 1) v += __shfl_xor(v, m, 32);
    sacc[jj] = v;
  }
  if ((lane & 15) == 0) {
    const int rbase = m0 + ((lane >> 4) << 3);
#pragma unroll
    for (int jj = 0; jj < 8; ++jj) outv[rbase + jj] = sacc[jj];
  }
}

// softmax over T=64 tokens for each (s,b); a layout [S][T][B]
__global__ void softmax_kernel(float* __restrict__ a) {
  __shared__ float red[64];
  const int sb = blockIdx.x, s = sb >> 6, b = sb & 63, t = threadIdx.x;
  const int idx = (s * 64 + t) * 64 + b;
  float v = a[idx];
  red[t] = v; __syncthreads();
  for (int st = 32; st > 0; st >>= 1) {
    if (t < st) red[t] = fmaxf(red[t], red[t + st]);
    __syncthreads();
  }
  const float mx = red[0]; __syncthreads();
  const float e = __expf(v - mx);
  red[t] = e; __syncthreads();
  for (int st = 32; st > 0; st >>= 1) {
    if (t < st) red[t] += red[t + st];
    __syncthreads();
  }
  a[idx] = e * __builtin_amdgcn_rcpf(red[0]);
}

// sent[s,b,h] = sum_t a1[s,t,b] * intra_out[s,t,b,h]   (bf16 out)
__global__ void sentvec_kernel(const float* __restrict__ a1,
                               const unsigned short* __restrict__ io,
                               unsigned short* __restrict__ sent) {
  const int i = blockIdx.x * 256 + threadIdx.x;   // S*B*256
  const int h = i & 255, sb = i >> 8, s = sb >> 6, b = sb & 63;
  float acc = 0.f;
  for (int t = 0; t < 64; ++t) {
    const int row = (s * 64 + t) * 64 + b;
    acc += a1[row] * b2f(io[(size_t)row * 256 + h]);
  }
  sent[i] = f2b(acc);
}

// doc_vec + final linear: out[b,c] = b_final[c] + sum_h doc[b,h]*W_final[c,h]
__global__ void final_kernel(const float* __restrict__ a2,
                             const unsigned short* __restrict__ io,
                             const float* __restrict__ Wf,
                             const float* __restrict__ bf,
                             float* __restrict__ outp) {
  __shared__ float dv[256];
  const int b = blockIdx.x, tid = threadIdx.x;
  float acc = 0.f;
  for (int s = 0; s < 32; ++s)
    acc += a2[s * 64 + b] * b2f(io[(size_t)(s * 64 + b) * 256 + tid]);
  dv[tid] = acc;
  __syncthreads();
  const int c = tid >> 5, lane = tid & 31;
  if (c < 5) {
    float p = 0.f;
    for (int h = lane; h < 256; h += 32) p += dv[h] * Wf[c * 256 + h];
    for (int m = 16; m >= 1; m >>= 1) p += __shfl_xor(p, m, 32);
    if (lane == 0) outp[b * 5 + c] = p + bf[c];
  }
}

// ---------------------------------------------------------------------------
extern "C" void kernel_launch(void* const* d_in, const int* in_sizes, int n_in,
                              void* d_out, int out_size, void* d_ws, size_t ws_size,
                              hipStream_t stream) {
  (void)in_sizes; (void)n_in; (void)out_size; (void)ws_size;
  const int*   tokens    = (const int*)  d_in[0];
  const float* embed     = (const float*)d_in[1];
  const float* Wih_intra = (const float*)d_in[2];
  const float* Whh_intra = (const float*)d_in[3];
  const float* bih_intra = (const float*)d_in[4];
  const float* bhh_intra = (const float*)d_in[5];
  const float* WW_intra  = (const float*)d_in[6];
  const float* b_intra   = (const float*)d_in[7];
  const float* p_intra   = (const float*)d_in[8];
  const float* Wih_inter = (const float*)d_in[9];
  const float* Whh_inter = (const float*)d_in[10];
  const float* bih_inter = (const float*)d_in[11];
  const float* bhh_inter = (const float*)d_in[12];
  const float* WW_inter  = (const float*)d_in[13];
  const float* b_inter   = (const float*)d_in[14];
  const float* p_inter   = (const float*)d_in[15];
  const float* W_final   = (const float*)d_in[16];
  const float* b_final   = (const float*)d_in[17];
  float* outp = (float*)d_out;

  char* p = (char*)d_ws;
  unsigned short* emb_b   = (unsigned short*)(p);              // 33,554,432 B
  unsigned short* intra_b = (unsigned short*)(p +  33554432);  // 67,108,864 B
  float*          a1      = (float*)         (p + 100663296);  //    524,288 B
  unsigned short* sent_b  = (unsigned short*)(p + 101187584);  //  1,048,576 B
  unsigned short* inter_b = (unsigned short*)(p + 102236160);  //  1,048,576 B
  float*          a2      = (float*)         (p + 103284736);  //      8,192 B
  unsigned short* WhhI_b  = (unsigned short*)(p + 103292928);  //    196,608 B
  unsigned short* WihI_b  = (unsigned short*)(p + 103489536);  //    196,608 B
  unsigned short* WhhG_b  = (unsigned short*)(p + 103686144);  //    196,608 B
  unsigned short* WihG_b  = (unsigned short*)(p + 103882752);  //    393,216 B
  unsigned short* WWiT_b  = (unsigned short*)(p + 104275968);  //    131,072 B
  unsigned short* WWgT_b  = (unsigned short*)(p + 104407040);  //    131,072 B

  // weight precision conversion / transposition
  cvt_kernel<<<dim3(98304 / 256),  dim3(256), 0, stream>>>(Whh_intra, WhhI_b, 98304);
  cvt_kernel<<<dim3(98304 / 256),  dim3(256), 0, stream>>>(Wih_intra, WihI_b, 98304);
  cvt_kernel<<<dim3(98304 / 256),  dim3(256), 0, stream>>>(Whh_inter, WhhG_b, 98304);
  cvt_kernel<<<dim3(196608 / 256), dim3(256), 0, stream>>>(Wih_inter, WihG_b, 196608);
  transp256_kernel<<<dim3(256), dim3(256), 0, stream>>>(WW_intra, WWiT_b);
  transp256_kernel<<<dim3(256), dim3(256), 0, stream>>>(WW_inter, WWgT_b);

  // embedding gather -> bf16 [S,T,B,E]  (8 elems/thread)
  embed_kernel<<<dim3(2097152 / 256), dim3(256), 0, stream>>>(tokens, embed, emb_b);

  // intra biGRU: 32 sentences x 2 directions
  gru_kernel<64, 128><<<dim3(32, 2), dim3(256), 0, stream>>>(
      emb_b, WhhI_b, WihI_b, bih_intra, bhh_intra, intra_b);

  // word attention logits + softmax over T + sentence vectors
  attn_kernel<<<dim3(131072 / 128), dim3(256), 0, stream>>>(
      intra_b, WWiT_b, b_intra, p_intra, a1, 131072);
  softmax_kernel<<<dim3(2048), dim3(64), 0, stream>>>(a1);
  sentvec_kernel<<<dim3(524288 / 256), dim3(256), 0, stream>>>(a1, intra_b, sent_b);

  // inter biGRU: 1 sequence of S=32 steps x 2 directions, input 2H=256
  gru_kernel<32, 256><<<dim3(1, 2), dim3(256), 0, stream>>>(
      sent_b, WhhG_b, WihG_b, bih_inter, bhh_inter, inter_b);

  // sentence attention (no softmax) + doc vector + final linear
  attn_kernel<<<dim3(2048 / 128), dim3(256), 0, stream>>>(
      inter_b, WWgT_b, b_inter, p_inter, a2, 2048);
  final_kernel<<<dim3(64), dim3(256), 0, stream>>>(a2, inter_b, W_final, b_final, outp);
}
